// DetectionLoss_72773925863459
// MI455X (gfx1250) — matrix operand order repaired and checked by reading, not compiled, not run on hardware
//
#include <hip/hip_runtime.h>
#include <hip/hip_bf16.h>

// ---------------------------------------------------------------------------
// DetectionLoss for MI455X (gfx1250, wave32, WMMA).
//   B=8 images, N=200 padded detections, pad to NPAD=208 (13 x 16-tiles).
//   ws layout: [ iou: B*NPAD*NPAD floats ][ nab: B*2 floats ]
//
// Pipeline:
//   zero_out -> count_valid (ballot/popc) -> iou_wmma (mask GEMM via
//   v_wmma_f32_16x16x32_f16 + VALU geometric IoU) -> hungarian_loss
//   (register-resident JV, one column per thread, wave32 shuffle argmin).
// ---------------------------------------------------------------------------

typedef __attribute__((ext_vector_type(16))) _Float16 v16h;
typedef __attribute__((ext_vector_type(8)))  float    v8f;

#define BATCH     8
#define NDET      200
#define NPAD      208
#define NTI       (NPAD / 16)      // 13 tiles per dimension
#define SCORE_THR 0.001f
#define HUGE_COST 1.0e30f

__device__ __forceinline__ bool allowed_cls(int c) {
    return (c == 1) | (c == 2) | (c == 3) | (c == 5) | (c == 7);
}

__device__ __forceinline__ bool box_valid(const float* p) {
    // p -> [x1,y1,x2,y2,score,cls]
    int c = (int)p[5];
    return (p[4] > SCORE_THR) && allowed_cls(c);
}

// Order-preserving float -> uint32 map (monotone for all finite floats),
// packed with column index so u64-min == (min value, then lowest index).
__device__ __forceinline__ unsigned long long pack_min(float v, int idx) {
    unsigned int u = __float_as_uint(v);
    u = (u & 0x80000000u) ? ~u : (u | 0x80000000u);
    return ((unsigned long long)u << 32) | (unsigned int)idx;
}

__device__ __forceinline__ float unpack_val(unsigned long long m) {
    unsigned int u = (unsigned int)(m >> 32);
    u = (u & 0x80000000u) ? (u & 0x7fffffffu) : ~u;
    return __uint_as_float(u);
}

__device__ __forceinline__ unsigned long long wave_min_u64(unsigned long long x) {
#pragma unroll
    for (int off = 16; off > 0; off >>= 1) {
        unsigned long long o = __shfl_xor(x, off, 32);
        if (o < x) x = o;
    }
    return x;
}

// ---------------------------------------------------------------------------
__global__ void zero_out_kernel(float* out) { out[0] = 0.0f; }

// ---------------------------------------------------------------------------
// Count valid boxes per image -> nab[b*2 + {0,1}]; wave ballot + popcount.
// ---------------------------------------------------------------------------
__global__ void count_valid_kernel(const float* __restrict__ b1,
                                   const float* __restrict__ b2,
                                   float* __restrict__ nab) {
    __shared__ int c1, c2;
    const int b = blockIdx.x, t = threadIdx.x;
    if (t == 0) { c1 = 0; c2 = 0; }
    __syncthreads();
    const bool in = t < NDET;
    const bool v1 = in && box_valid(b1 + ((size_t)b * NDET + t) * 6);
    const bool v2 = in && box_valid(b2 + ((size_t)b * NDET + t) * 6);
    const unsigned long long m1 = __ballot(v1);
    const unsigned long long m2 = __ballot(v2);
    if ((t & 31) == 0) {
        atomicAdd(&c1, __popcll(m1));
        atomicAdd(&c2, __popcll(m2));
    }
    __syncthreads();
    if (t == 0) { nab[2 * b + 0] = (float)c1; nab[2 * b + 1] = (float)c2; }
}

// ---------------------------------------------------------------------------
// Masked IoU via WMMA: mask[i][j] = <onehot(cls1_i)*valid1_i,
// onehot(cls2_j)*valid2_j> as one V_WMMA_F32_16X16X32_F16 per 16x16 tile
// (K=32 >= 10 classes). One wave32 per tile; geometric IoU where mask==1.
// ---------------------------------------------------------------------------
__global__ void iou_wmma_kernel(const float* __restrict__ b1,
                                const float* __restrict__ b2,
                                float* __restrict__ iou) {
    const int blk  = blockIdx.x;
    const int b    = blk / (NTI * NTI);
    const int rem  = blk % (NTI * NTI);
    const int tm   = rem / NTI;          // tile row (boxes1)
    const int tn   = rem % NTI;          // tile col (boxes2)
    const int lane = threadIdx.x;        // 0..31
    const bool hi  = lane >= 16;

    // A fragment: 16x32 f16 (rows = boxes1). Lanes 0-15: M=lane;
    // VGPR v<4: K={0..7 | 8..15(hi)} pairs, v>=4: K={16..23 | 24..31(hi)}.
    const int arow = tm * 16 + (lane & 15);
    int acls = -1;
    if (arow < NDET) {
        const float* p = b1 + ((size_t)b * NDET + arow) * 6;
        if (box_valid(p)) acls = (int)p[5];
    }
    v16h afrag;
#pragma unroll
    for (int e = 0; e < 16; ++e) {
        const int v = e >> 1, h = e & 1;
        const int K = (v < 4) ? ((hi ? 8 : 0) + 2 * v + h)
                              : ((hi ? 24 : 16) + 2 * (v - 4) + h);
        afrag[e] = (acls == K) ? (_Float16)1.0f : (_Float16)0.0f;
    }

    // B fragment: 32x16 f16 (cols = boxes2). N=lane&15; e<8: K=(hi?8:0)+e;
    // e>=8: K=(hi?24:16)+(e-8).
    const int bcol = tn * 16 + (lane & 15);
    int bcls = -1;
    if (bcol < NDET) {
        const float* p = b2 + ((size_t)b * NDET + bcol) * 6;
        if (box_valid(p)) bcls = (int)p[5];
    }
    v16h bfrag;
#pragma unroll
    for (int e = 0; e < 16; ++e) {
        const int K = (e < 8) ? ((hi ? 8 : 0) + e) : ((hi ? 24 : 16) + (e - 8));
        bfrag[e] = (bcls == K) ? (_Float16)1.0f : (_Float16)0.0f;
    }

    v8f c = {0.f, 0.f, 0.f, 0.f, 0.f, 0.f, 0.f, 0.f};
    c = __builtin_amdgcn_wmma_f32_16x16x32_f16(
        false, afrag, false, bfrag, (short)0, c, false, false);

    // C/D layout: VGPR r -> M = (hi?8:0)+r ; N = lane&15.
    const int gj = tn * 16 + (lane & 15);
#pragma unroll
    for (int r = 0; r < 8; ++r) {
        const int gi = tm * 16 + (hi ? 8 : 0) + r;
        float val = 0.0f;
        if (gi < NDET && gj < NDET && c[r] > 0.5f) {
            const float* p1 = b1 + ((size_t)b * NDET + gi) * 6;
            const float* p2 = b2 + ((size_t)b * NDET + gj) * 6;
            const float x1 = fmaxf(p1[0], p2[0]);
            const float y1 = fmaxf(p1[1], p2[1]);
            const float x2 = fminf(p1[2], p2[2]);
            const float y2 = fminf(p1[3], p2[3]);
            const float inter = fmaxf(x2 - x1, 0.0f) * fmaxf(y2 - y1, 0.0f);
            const float a1 = (p1[2] - p1[0]) * (p1[3] - p1[1]);
            const float a2 = (p2[2] - p2[0]) * (p2[3] - p2[1]);
            const float uni = a1 + a2 - inter;
            if (uni > 0.0f) val = inter * __builtin_amdgcn_rcpf(uni);
        }
        iou[((size_t)b * NPAD + gi) * NPAD + gj] = val;   // padding written as 0
    }
}

// ---------------------------------------------------------------------------
// Jonker-Volgenant shortest augmenting path, one workgroup (256 thr = 8 wave32)
// per image. 1-based (NPAD+1) system; a(i,j) = -iou[i-1][j-1], a(0,*)=a(*,0)=0.
// Each thread OWNS one column j==tid: v[j], minv[j], used[j] live in registers;
// argmin = wave32 shuffle reduction + 8-partial LDS combine (3 barriers/iter).
// ---------------------------------------------------------------------------
__global__ void hungarian_loss_kernel(const float* __restrict__ iou,
                                      const float* __restrict__ nab,
                                      float* __restrict__ out) {
    const int b   = blockIdx.x;
    const int tid = threadIdx.x;
    const int n   = NPAD;                              // 208
    const float* A = iou + (size_t)b * NPAD * NPAD;
    const bool owner = (tid <= n);                     // threads 0..208 own a column

    __shared__ float u_sh[NPAD + 1];                   // row potentials
    __shared__ int   p_sh[NPAD + 1];                   // p[j] = row matched to col j
    __shared__ int   way_sh[NPAD + 1];
    __shared__ unsigned long long part[8];
    __shared__ unsigned long long bcast;
    __shared__ float fpart[8];

    if (owner) { u_sh[tid] = 0.0f; p_sh[tid] = 0; way_sh[tid] = 0; }
    float vr = 0.0f;                                   // column potential (register)
    __syncthreads();

    for (int i = 1; i <= n; ++i) {
        if (tid == 0) {
            p_sh[0] = i;
            __builtin_prefetch(A + (size_t)(i - 1) * NPAD, 0, 0);  // global_prefetch_b8
        }
        float minvr = HUGE_COST;                       // register minv[tid]
        int   usedr = 0;                               // register used[tid]
        __syncthreads();

        int   j0  = 0;
        int   i0  = i;                                 // p_sh[0]
        float ui0 = u_sh[i0];

        while (true) {
            if (tid == j0) usedr = 1;

            // relaxation of this thread's column against row i0
            float cand = HUGE_COST;
            if (owner && !usedr) {
                const float cost = (tid >= 1 && i0 >= 1)
                                 ? -A[(size_t)(i0 - 1) * NPAD + (tid - 1)] : 0.0f;
                const float cur = cost - ui0 - vr;
                if (cur < minvr) { minvr = cur; way_sh[tid] = j0; }
                cand = minvr;
            }

            // block-wide argmin: wave32 shuffles, then 8 partials in LDS
            unsigned long long pk = pack_min(cand, owner ? tid : NPAD);
            pk = wave_min_u64(pk);
            if ((tid & 31) == 0) part[tid >> 5] = pk;
            __syncthreads();
            if (tid < 8) {
                unsigned long long q = part[tid];
#pragma unroll
                for (int off = 4; off > 0; off >>= 1) {
                    unsigned long long o = __shfl_xor(q, off, 32);
                    if (o < q) q = o;
                }
                if (tid == 0) bcast = q;
            }
            __syncthreads();
            const unsigned long long m = bcast;
            const int   jn    = (int)(unsigned int)(m & 0xffffffffu);
            const float delta = unpack_val(m);

            // dual update: used columns' matched rows are distinct -> race-free
            if (owner) {
                if (usedr) { u_sh[p_sh[tid]] += delta; vr -= delta; }
                else       { minvr -= delta; }
            }
            j0 = jn;
            const int i0n = p_sh[jn];                  // p unchanged inside path
            if (i0n == 0) break;                       // free column reached
            i0 = i0n;
            __syncthreads();                           // u writes visible; part[] reuse safe
            ui0 = u_sh[i0];
        }

        __syncthreads();                               // way_sh visible to tid 0
        if (tid == 0) {                                // augment along `way` chain
            int j = j0;
            while (j != 0) { const int j1 = way_sh[j]; p_sh[j] = p_sh[j1]; j = j1; }
        }
        __syncthreads();
    }

    // matched IoU sum: column tid matched to row p_sh[tid]
    float s = 0.0f;
    if (tid >= 1 && tid <= n) {
        const int ii = p_sh[tid];
        if (ii >= 1) s = A[(size_t)(ii - 1) * NPAD + (tid - 1)];
    }
#pragma unroll
    for (int off = 16; off > 0; off >>= 1) s += __shfl_xor(s, off, 32);
    if ((tid & 31) == 0) fpart[tid >> 5] = s;
    __syncthreads();

    if (tid == 0) {
        float tot = 0.0f;
#pragma unroll
        for (int k = 0; k < 8; ++k) tot += fpart[k];
        const float na = nab[2 * b + 0], nb = nab[2 * b + 1];
        const float denom = fmaxf(fmaxf(na, nb), 1.0f);
        float per = 1.0f - tot / denom;
        const bool az = (na == 0.0f), bz = (nb == 0.0f);
        per = (az && bz) ? 0.0f : ((az != bz) ? 1.0f : per);
        atomicAdd(out, per / (float)BATCH);
    }
}

// ---------------------------------------------------------------------------
extern "C" void kernel_launch(void* const* d_in, const int* in_sizes, int n_in,
                              void* d_out, int out_size, void* d_ws, size_t ws_size,
                              hipStream_t stream) {
    const float* boxes1 = (const float*)d_in[0];   // [B, N, 6] f32
    const float* boxes2 = (const float*)d_in[1];   // [B, N, 6] f32
    float* iou = (float*)d_ws;                               // B*NPAD*NPAD
    float* nab = iou + (size_t)BATCH * NPAD * NPAD;          // B*2
    float* out = (float*)d_out;                              // scalar loss

    zero_out_kernel<<<1, 1, 0, stream>>>(out);
    count_valid_kernel<<<BATCH, 256, 0, stream>>>(boxes1, boxes2, nab);
    iou_wmma_kernel<<<BATCH * NTI * NTI, 32, 0, stream>>>(boxes1, boxes2, iou);
    hungarian_loss_kernel<<<BATCH, 256, 0, stream>>>(iou, nab, out);
}
